// RNN_56581899157698
// MI455X (gfx1250) — matrix operand order repaired and checked by reading, not compiled, not run on hardware
//
#include <hip/hip_runtime.h>

#define SEQ_T 2048
#define BATCH 256
#define DIN   64
#define H     128

typedef __attribute__((ext_vector_type(16))) __bf16 v16bf;
typedef __attribute__((ext_vector_type(8)))  float  v8f;
typedef __attribute__((ext_vector_type(4)))  float  vf4;

union FragBF {
  v16bf v;
  __bf16 h[16];
  uint4 q[2];
};

__device__ __forceinline__ unsigned short bf1(float a) {
  union { __bf16 h; unsigned short s; } r;
  r.h = (__bf16)a;
  return r.s;
}

__device__ __forceinline__ unsigned int bf2(float a, float b) {
  union { __bf16 h[2]; unsigned int u; } r;
  r.h[0] = (__bf16)a; r.h[1] = (__bf16)b;
  return r.u;
}

// tanh-form GELU: one hardware transcendental, branchless
__device__ __forceinline__ float gelu_fast(float x) {
  float z = 0.79788456080286536f * (x + 0.044715f * x * x * x);
#if __has_builtin(__builtin_amdgcn_tanhf)
  float th = __builtin_amdgcn_tanhf(z);          // v_tanh_f32
#else
  // tanh(z) = 1 - 2/(exp(2z)+1); exp(2z) = exp2(z * 2*log2(e))
  float e  = __builtin_amdgcn_exp2f(z * 2.885390081777926815f);
  float th = 1.0f - 2.0f * __builtin_amdgcn_rcpf(e + 1.0f);
#endif
  return 0.5f * x * (1.0f + th);
}

// A-matrix 16x32 bf16 fragment from LDS (row-major, rowStride in ushorts).
// ISA 7.12.2: lane m=lane%16; regs hold K = kc + laneHi*8 + {0..7, 16..23}.
__device__ __forceinline__ v16bf load_a_frag(const unsigned short* sh, int rowStride,
                                             int m, int kc, int laneHi) {
  FragBF f;
  f.q[0] = *(const uint4*)(sh + m * rowStride + kc +      laneHi * 8);
  f.q[1] = *(const uint4*)(sh + m * rowStride + kc + 16 + laneHi * 8);
  return f.v;
}

// B-matrix 32x16 bf16 fragment (B = W^T, W row-major [rows][K]).
// Lane holds column n (= row n of W): K = kc + laneHi*8 + {0..7, 16..23}.
__device__ __forceinline__ v16bf load_b_frag(const float* __restrict__ W, int K,
                                             int n, int kc, int laneHi) {
  FragBF f;
  const float* p = W + (size_t)n * K + kc + laneHi * 8;
#pragma unroll
  for (int j = 0; j < 16; ++j) f.h[j] = (__bf16)p[j + (j >= 8 ? 8 : 0)];
  return f.v;
}

#define WMMA_BF16(A, B, C) \
  __builtin_amdgcn_wmma_f32_16x16x32_bf16(false, (A), false, (B), (short)0, (C), false, false)

__global__ __launch_bounds__(256, 1)
void rnn2_fused(const float* __restrict__ seq,
                const float* __restrict__ Win0, const float* __restrict__ Wh0,
                const float* __restrict__ Win1, const float* __restrict__ Wh1,
                const float* __restrict__ h00,  const float* __restrict__ h01,
                float* __restrict__ out) {
  // double-buffered tiles: 2 barriers/step instead of 4
  __shared__ __align__(16) unsigned short sh_x [2][16 * DIN];
  __shared__ __align__(16) unsigned short sh_h0[2][16 * H];
  __shared__ __align__(16) unsigned short sh_h1[2][16 * H];

  const int tid    = threadIdx.x;
  const int lane   = tid & 31;
  const int wave   = tid >> 5;        // 0..7 -> N-tile
  const int laneHi = lane >> 4;       // 0 | 1
  const int lmod   = lane & 15;
  const int b0     = blockIdx.x * 16; // 16 batch rows per WG

  // init hidden tiles (buffer 0) from learned h0 params
  for (int i = tid; i < 16 * H; i += 256) {
    const int c = i & (H - 1);
    sh_h0[0][i] = bf1(h00[c]);
    sh_h1[0][i] = bf1(h01[c]);
  }

  // per-wave weight B-fragments in registers (wave owns cols nb..nb+15)
  const int nb = wave * 16;
  const int n  = nb + lmod;
  v16bf bWin0[2], bWh0[4], bWin1[4], bWh1[4];
#pragma unroll
  for (int c = 0; c < 2; ++c) bWin0[c] = load_b_frag(Win0, DIN, n, c * 32, laneHi);
#pragma unroll
  for (int c = 0; c < 4; ++c) {
    bWh0[c]  = load_b_frag(Wh0,  H, n, c * 32, laneHi);
    bWin1[c] = load_b_frag(Win1, H, n, c * 32, laneHi);
    bWh1[c]  = load_b_frag(Wh1,  H, n, c * 32, laneHi);
  }

  // stage x[0] into buffer 0 (tile is one contiguous 1024-float run)
  {
    vf4 v = __builtin_nontemporal_load(((const vf4*)(seq + (size_t)b0 * DIN)) + tid);
    *(uint2*)(&sh_x[0][tid * 4]) = make_uint2(bf2(v.x, v.y), bf2(v.z, v.w));
  }
  __syncthreads();

  float g0[8], g1[8];  // last-step activations (final hidden states)

#pragma unroll 1
  for (int t = 0; t < SEQ_T; ++t) {
    const int p = t & 1, q = p ^ 1;

    // 1) issue global load of x[t+1] early; LDS store deferred to end of step
    vf4 xv;
    if (t + 1 < SEQ_T) {
      const vf4* src = ((const vf4*)(seq + ((size_t)(t + 1) * BATCH + b0) * DIN)) + tid;
      xv = __builtin_nontemporal_load(src);
      if (t + 2 < SEQ_T)
        __builtin_prefetch((const void*)(src + (size_t)BATCH * DIN / 4), 0, 1);
    }

    // 2) layer 0: acc0 = x[t] @ Win0^T + h0 @ Wh0^T  (two independent chains)
    v8f acc0;
    {
      v16bf a0 = load_a_frag(sh_x[p],  DIN, lmod,  0, laneHi);
      v16bf a1 = load_a_frag(sh_x[p],  DIN, lmod, 32, laneHi);
      v16bf a2 = load_a_frag(sh_h0[p], H, lmod,  0, laneHi);
      v16bf a3 = load_a_frag(sh_h0[p], H, lmod, 32, laneHi);
      v16bf a4 = load_a_frag(sh_h0[p], H, lmod, 64, laneHi);
      v16bf a5 = load_a_frag(sh_h0[p], H, lmod, 96, laneHi);
      v8f cA = {}, cB = {};
      cA = WMMA_BF16(a0, bWin0[0], cA);  cB = WMMA_BF16(a1, bWin0[1], cB);
      cA = WMMA_BF16(a2, bWh0[0],  cA);  cB = WMMA_BF16(a3, bWh0[1],  cB);
      cA = WMMA_BF16(a4, bWh0[2],  cA);  cB = WMMA_BF16(a5, bWh0[3],  cB);
      acc0 = cA + cB;
    }

    // 3) h0 <- gelu(acc0) into buffer q (no WAR: different buffer)
#pragma unroll
    for (int r = 0; r < 8; ++r) {
      const int m = r + laneHi * 8;
      const float g = gelu_fast(acc0[r]);
      g0[r] = g;
      sh_h0[q][m * H + nb + lmod] = bf1(g);
    }
    __syncthreads();  // h0[q] visible to all waves

    // 4) layer 1: acc1 = y0 @ Win1^T + h1 @ Wh1^T
    v8f acc1;
    {
      v8f cA = {}, cB = {};
      {
        v16bf a0 = load_a_frag(sh_h0[q], H, lmod,  0, laneHi);
        v16bf a1 = load_a_frag(sh_h0[q], H, lmod, 32, laneHi);
        v16bf a2 = load_a_frag(sh_h0[q], H, lmod, 64, laneHi);
        v16bf a3 = load_a_frag(sh_h0[q], H, lmod, 96, laneHi);
        cA = WMMA_BF16(a0, bWin1[0], cA);  cB = WMMA_BF16(a1, bWin1[1], cB);
        cA = WMMA_BF16(a2, bWin1[2], cA);  cB = WMMA_BF16(a3, bWin1[3], cB);
      }
      {
        v16bf a0 = load_a_frag(sh_h1[p], H, lmod,  0, laneHi);
        v16bf a1 = load_a_frag(sh_h1[p], H, lmod, 32, laneHi);
        v16bf a2 = load_a_frag(sh_h1[p], H, lmod, 64, laneHi);
        v16bf a3 = load_a_frag(sh_h1[p], H, lmod, 96, laneHi);
        cA = WMMA_BF16(a0, bWh1[0], cA);  cB = WMMA_BF16(a1, bWh1[1], cB);
        cA = WMMA_BF16(a2, bWh1[2], cA);  cB = WMMA_BF16(a3, bWh1[3], cB);
      }
      acc1 = cA + cB;
    }

    // 5) y1[t] = gelu(acc1): streaming store to global, new h1 into buffer q
#pragma unroll
    for (int r = 0; r < 8; ++r) {
      const int m = r + laneHi * 8;
      const float g = gelu_fast(acc1[r]);
      g1[r] = g;
      __builtin_nontemporal_store(g, &out[((size_t)t * BATCH + b0 + m) * H + nb + lmod]);
      sh_h1[q][m * H + nb + lmod] = bf1(g);
    }

    // 6) park staged x[t+1] into buffer q
    if (t + 1 < SEQ_T)
      *(uint2*)(&sh_x[q][tid * 4]) = make_uint2(bf2(xv.x, xv.y), bf2(xv.z, xv.w));

    __syncthreads();  // h1[q], x[q] visible; also guards all WAR for next step
  }

  // final hiddens: out[T*B*H + layer*B*H + b*H + h]
  const size_t fh = (size_t)SEQ_T * BATCH * H;
#pragma unroll
  for (int r = 0; r < 8; ++r) {
    const int m = r + laneHi * 8;
    out[fh + (size_t)(b0 + m) * H + nb + lmod]                     = g0[r];
    out[fh + (size_t)BATCH * H + (size_t)(b0 + m) * H + nb + lmod] = g1[r];
  }
}

extern "C" void kernel_launch(void* const* d_in, const int* in_sizes, int n_in,
                              void* d_out, int out_size, void* d_ws, size_t ws_size,
                              hipStream_t stream) {
  (void)in_sizes; (void)n_in; (void)out_size; (void)d_ws; (void)ws_size;
  const float* seq  = (const float*)d_in[0];
  const float* Win0 = (const float*)d_in[1];
  const float* Wh0  = (const float*)d_in[2];
  const float* Win1 = (const float*)d_in[3];
  const float* Wh1  = (const float*)d_in[4];
  const float* h00  = (const float*)d_in[5];
  const float* h01  = (const float*)d_in[6];
  float* out = (float*)d_out;

  rnn2_fused<<<dim3(BATCH / 16), dim3(256), 0, stream>>>(
      seq, Win0, Wh0, Win1, Wh1, h00, h01, out);
}
